// UResNet_35948876268435
// MI455X (gfx1250) — matrix operand order repaired and checked by reading, not compiled, not run on hardware
//
#include <hip/hip_runtime.h>
#include <hip/hip_bf16.h>
#include <stdint.h>
#include <stddef.h>

// ---------------------------------------------------------------------------
// Sparse UResNet on MI455X (gfx1250).
// Working set is L2-resident (192MB), so the win is moving channel
// contractions onto the WMMA pipes (bf16 operands, f32 accumulate) and doing
// the sparse gathers as 16-byte vector loads staged through LDS.
// ---------------------------------------------------------------------------

#define BN_EPS 1e-4f

typedef __bf16 bf16;
typedef __attribute__((ext_vector_type(16))) __bf16 v16bf;
typedef __attribute__((ext_vector_type(8)))  __bf16 v8bf;
typedef __attribute__((ext_vector_type(8)))  float  v8f;

static inline int cdiv(int a, int b) { return (a + b - 1) / b; }

// ---------------------------------------------------------------------------
// Elementwise f32 -> bf16
// ---------------------------------------------------------------------------
__global__ void k_f32_to_bf16(const float* __restrict__ x, bf16* __restrict__ y, int n) {
    int i = blockIdx.x * blockDim.x + threadIdx.x;
    if (i < n) y[i] = (bf16)x[i];
}

// Weight convert + transpose: W[k][d] (row-major, Ktot x cout) -> Wt[d][k]
__global__ void k_w_to_bf16_t(const float* __restrict__ W, bf16* __restrict__ Wt,
                              int Ktot, int cout) {
    int i = blockIdx.x * blockDim.x + threadIdx.x;
    if (i >= Ktot * cout) return;
    int k = i / cout, d = i - k * cout;
    Wt[(size_t)d * Ktot + k] = (bf16)W[i];
}

// Fuse rulebook: out[i] = mask[i] != 0 ? nbr[i] : -1  (one load/run in convs)
__global__ void k_fuse_rule(const int* __restrict__ nbr, const float* __restrict__ mask,
                            int* __restrict__ out, int n) {
    int i = blockIdx.x * blockDim.x + threadIdx.x;
    if (i < n) out[i] = (mask[i] != 0.f) ? nbr[i] : -1;
}

// ---------------------------------------------------------------------------
// Input conv (cin == 1): out[n,d] = sum_k mask[k,n] * feats[nbr[k,n]] * W[k,d]
// Tiny (27 MAC per output); plain VALU kernel, f32 weights.
// ---------------------------------------------------------------------------
__global__ void k_in_conv(const float* __restrict__ feats, const float* __restrict__ W,
                          const int* __restrict__ nbrm,
                          float* __restrict__ out, int N) {
    int i = blockIdx.x * blockDim.x + threadIdx.x;
    if (i >= N * 16) return;
    int n = i >> 4, d = i & 15;
    float s = 0.f;
    for (int k = 0; k < 27; ++k) {
        int idx = nbrm[(size_t)k * N + n];
        if (idx >= 0)
            s += feats[idx] * W[k * 16 + d];
    }
    out[i] = s;
}

// ---------------------------------------------------------------------------
// Per-channel sum / sum-of-squares (one block per channel).
// ---------------------------------------------------------------------------
__global__ void k_colsum(const float* __restrict__ x, int n, int c,
                         float* __restrict__ sums, float* __restrict__ sqs) {
    __shared__ float ss[256];
    __shared__ float sq[256];
    int j = blockIdx.x;
    int t = threadIdx.x;
    float s = 0.f, q = 0.f;
    for (int i = t; i < n; i += 256) {
        float v = x[(size_t)i * c + j];
        s += v; q += v * v;
    }
    ss[t] = s; sq[t] = q;
    __syncthreads();
    for (int o = 128; o > 0; o >>= 1) {
        if (t < o) { ss[t] += ss[t + o]; sq[t] += sq[t + o]; }
        __syncthreads();
    }
    if (t == 0) { sums[j] = ss[0]; sqs[j] = sq[0]; }
}

// ---------------------------------------------------------------------------
// Fused batch-norm (batch stats) + ReLU + cast to bf16.
// ---------------------------------------------------------------------------
__global__ void k_bn_relu_bf16(const float* __restrict__ x,
                               const float* __restrict__ sums,
                               const float* __restrict__ sqs,
                               const float* __restrict__ g,
                               const float* __restrict__ b,
                               int n, int c, bf16* __restrict__ out) {
    int i = blockIdx.x * blockDim.x + threadIdx.x;
    if (i >= n * c) return;
    int j = i % c;
    float inv_n = 1.0f / (float)n;
    float mu  = sums[j] * inv_n;
    float var = sqs[j] * inv_n - mu * mu;
    float v = (x[i] - mu) * rsqrtf(var + BN_EPS) * g[j] + b[j];
    out[i] = (bf16)(v > 0.f ? v : 0.f);
}

// ---------------------------------------------------------------------------
// Fused gather-GEMM on WMMA.  REQUIRES cin % 8 == 0.
//   out[n, colOff+j] (+)= sum_{k,c} X[nbrm[k,n], c] * W[k*cin+c, coutBase+j]
// One wave owns a 16(points) x 16(cout) tile. K = Ktot is consumed in 32-wide
// chunks with V_WMMA_F32_16X16X32_BF16.
//   A: per-wave LDS tile As[m][K] gathered with 16B global loads (cin % 8 == 0
//      guarantees an 8-K run never crosses a kernel-offset boundary).
//      kernel-offset index kk = (q * mul16) >> 16 (exact magic div by cin/8).
//   B: double-buffered block LDS tile Bs[pb][n][K] (N-major) from
//      pre-transposed weights Wt[d][k], loaded as 16B runs.
// Double-buffered B + wave-private A -> one __syncthreads per chunk.
// Fragments are four ds_load_b128 per lane, matching the ISA VGPR layout:
//   A lane: M = lane&15, K-runs {Kb..Kb+7, Kb+16..Kb+23}, Kb = (lane<16)?0:8
//   B lane: N = lane&15, K-runs {Kb..Kb+7, Kb+16..Kb+23}, Kb = (lane<16)?0:8
// nbrm==nullptr -> identity gather (dense GEMM / NiN). nbrm entries < 0 -> 0.
// ---------------------------------------------------------------------------
__global__ void __launch_bounds__(128)
k_gather_wmma(const bf16* __restrict__ X, const bf16* __restrict__ Wt,
              const int* __restrict__ nbrm,
              float* __restrict__ out, int Nout, int cin, int cin8,
              unsigned mul16, int Ktot, int ldc, int colOff, int accum) {
    __shared__ __align__(16) bf16 As[4][16 * 32];      // per-wave A tile [m][K]
    __shared__ __align__(16) bf16 Bs[2][16 * 32];      // double-buffered B [n][K]

    const int tid  = threadIdx.x;
    const int lane = tid & 31;
    const int wave = tid >> 5;
    const int rowBase  = (blockIdx.x * 4 + wave) * 16;
    const int coutBase = blockIdx.y * 16;

    const int mA  = lane & 15;
    const int nB  = lane & 15;
    const int KbA = (lane < 16) ? 0 : 8;
    const int KbB = (lane < 16) ? 0 : 8;

    v8f acc = {};
    const int nchunks = (Ktot + 31) >> 5;

#pragma unroll 2
    for (int ic = 0; ic < nchunks; ++ic) {
        const int kc = ic * 32;
        bf16* bsb = Bs[ic & 1];

        // ---- stage B: Bs[n][k] = Wt[(coutBase+n)*Ktot + kc+k], 16B runs ----
        if (tid < 64) {
            int Nn = tid & 15, oct = tid >> 4;
            int kg0 = kc + oct * 8;
            v8bf bv = {};
            if (kg0 < Ktot)
                bv = *(const v8bf*)(Wt + (size_t)(coutBase + Nn) * Ktot + kg0);
            *(v8bf*)(bsb + Nn * 32 + oct * 8) = bv;
            if (kg0 + 32 < Ktot)   // prefetch next chunk of this weight row
                __builtin_prefetch(Wt + (size_t)(coutBase + Nn) * Ktot + kg0 + 32, 0, 0);
        }

        // ---- stage A (wave-cooperative): run r = lane*2+j -> (m=r>>2, oct=r&3)
#pragma unroll
        for (int j = 0; j < 2; ++j) {
            int r = lane * 2 + j;
            int m = r >> 2, oct = r & 3;
            int nrow = rowBase + m;
            int kg0 = kc + oct * 8;
            v8bf av = {};
            if (nrow < Nout && kg0 < Ktot) {
                int q  = kg0 >> 3;                       // 8-run index
                int kk = (int)((q * mul16) >> 16);       // = q / cin8 (exact)
                int c0 = (q - kk * cin8) * 8;
                int idxr = nbrm ? nbrm[(size_t)kk * Nout + nrow] : nrow;
                if (idxr >= 0)
                    av = *(const v8bf*)(X + (size_t)idxr * cin + c0);
            }
            *(v8bf*)(As[wave] + m * 32 + oct * 8) = av;
        }
        __syncthreads();

        // ---- fragments: four 16B LDS loads per lane ----
        v8bf a_lo = *(const v8bf*)(As[wave] + mA * 32 + KbA);
        v8bf a_hi = *(const v8bf*)(As[wave] + mA * 32 + KbA + 16);
        v8bf b_lo = *(const v8bf*)(bsb + nB * 32 + KbB);
        v8bf b_hi = *(const v8bf*)(bsb + nB * 32 + KbB + 16);
        v16bf a = __builtin_shufflevector(a_lo, a_hi,
                    0, 1, 2, 3, 4, 5, 6, 7, 8, 9, 10, 11, 12, 13, 14, 15);
        v16bf b = __builtin_shufflevector(b_lo, b_hi,
                    0, 1, 2, 3, 4, 5, 6, 7, 8, 9, 10, 11, 12, 13, 14, 15);

        acc = __builtin_amdgcn_wmma_f32_16x16x32_bf16(
            /*neg_a=*/false, a, /*neg_b=*/false, b,
            /*c_mod=*/(short)0, acc, /*reuse_a=*/false, /*reuse_b=*/false);
    }

    // ---- scatter C tile (VGPR r -> M row, lane -> N col) ----
#pragma unroll
    for (int r = 0; r < 8; ++r) {
        int Mr = r + ((lane < 16) ? 0 : 8);
        int n  = rowBase + Mr;
        if (n < Nout) {
            size_t o = (size_t)n * ldc + colOff + nB;
            out[o] = accum ? (out[o] + acc[r]) : acc[r];
        }
    }
}

// ---------------------------------------------------------------------------
// Deconv: out[n,d] = sum_c Y[parent[n],c] * Wup[oi(n), c, d]
// ---------------------------------------------------------------------------
__global__ void k_deconv(const bf16* __restrict__ Y, const float* __restrict__ Wup,
                         const int* __restrict__ parent, const float* __restrict__ onehot,
                         float* __restrict__ out, int N, int Cc, int Cl,
                         int ldc, int colOff) {
    int i = blockIdx.x * blockDim.x + threadIdx.x;
    if (i >= N * Cl) return;
    int n = i / Cl, d = i - n * Cl;
    int oi = 0;
#pragma unroll
    for (int k = 1; k < 8; ++k)
        if (onehot[n * 8 + k] > 0.5f) oi = k;
    float s = 0.f;
    int p = parent[n];
    if (p >= 0) {
        const bf16*  yr = Y + (size_t)p * Cc;
        const float* wr = Wup + (size_t)oi * Cc * Cl + d;
        for (int c = 0; c < Cc; ++c) s += (float)yr[c] * wr[(size_t)c * Cl];
    }
    out[(size_t)n * ldc + colOff + d] = s;
}

// ---------------------------------------------------------------------------
// Strided column copy (also plain copy when lds==ldd, colOff==0).
// ---------------------------------------------------------------------------
__global__ void k_copy_cols(const float* __restrict__ src, float* __restrict__ dst,
                            int N, int C, int lds, int ldd, int colOff) {
    int i = blockIdx.x * blockDim.x + threadIdx.x;
    if (i >= N * C) return;
    int n = i / C, c = i - n * C;
    dst[(size_t)n * ldd + colOff + c] = src[(size_t)n * lds + c];
}

// Final linear head: (N x 16) @ (16 x 5) + b
__global__ void k_linear(const bf16* __restrict__ X, const float* __restrict__ Wl,
                         const float* __restrict__ bl, float* __restrict__ out,
                         int N, int C, int K) {
    int i = blockIdx.x * blockDim.x + threadIdx.x;
    if (i >= N * K) return;
    int n = i / K, k = i - n * K;
    float s = bl[k];
    for (int c = 0; c < C; ++c) s += (float)X[(size_t)n * C + c] * Wl[(size_t)c * K + k];
    out[i] = s;
}

// ---------------------------------------------------------------------------
// Host orchestration
// ---------------------------------------------------------------------------
struct BN    { const float* g; const float* b; };
struct Block { BN bn1, bn2; const float* conv1; const float* conv2; const float* nin; };

extern "C" void kernel_launch(void* const* d_in, const int* in_sizes, int n_in,
                              void* d_out, int out_size, void* d_ws, size_t ws_size,
                              hipStream_t stream) {
    (void)n_in; (void)out_size; (void)ws_size;

    auto f32p = [&](int i) { return (const float*)d_in[i]; };
    auto i32p = [&](int i) { return (const int*)d_in[i]; };

    // ---- parse inputs in jax pytree flatten order (sorted dict keys) ----
    int idx = 0;
    const float* feats = f32p(idx++);                       // 'feats'
    BN final_bn { f32p(idx), f32p(idx + 1) }; idx += 2;     // params.final_bn (g,b)
    const float* in_conv = f32p(idx++);                     // params.in_conv (27,1,16)

    const int DEPTH = 5, M16 = 16;
    int NP[5];
    for (int l = 0; l < DEPTH; ++l) NP[l] = M16 * (l + 1);  // [16,32,48,64,80]

    Block pre_b[5][2];
    Block post_b[4][2];
    BN down_bn[4], up_bn[4];
    const float* down_w[4];
    const float* up_w[4];

    auto readBlock = [&](bool has_nin) -> Block {
        Block bk;
        bk.bn1 = { f32p(idx), f32p(idx + 1) }; idx += 2;    // 'bn1'
        bk.bn2 = { f32p(idx), f32p(idx + 1) }; idx += 2;    // 'bn2'
        bk.conv1 = f32p(idx++);                             // 'conv1'
        bk.conv2 = f32p(idx++);                             // 'conv2'
        bk.nin = has_nin ? f32p(idx++) : nullptr;           // 'nin'
        return bk;
    };

    for (int l = 0; l < DEPTH; ++l) {
        if (l < DEPTH - 1) {
            // level keys sorted: down_bn, down_w, post, pre, up_bn, up_w
            down_bn[l] = { f32p(idx), f32p(idx + 1) }; idx += 2;
            down_w[l]  = f32p(idx++);
            post_b[l][0] = readBlock(true);   // cin=2cc != cout=cc -> nin present
            post_b[l][1] = readBlock(false);
            pre_b[l][0]  = readBlock(false);
            pre_b[l][1]  = readBlock(false);
            up_bn[l] = { f32p(idx), f32p(idx + 1) }; idx += 2;
            up_w[l]  = f32p(idx++);
        } else {
            pre_b[l][0] = readBlock(false);
            pre_b[l][1] = readBlock(false);
        }
    }
    const float* lin_b = f32p(idx++);
    const float* lin_w = f32p(idx++);

    // rules keys sorted: down_mask, down_nbr, mask, nbr, onehot, parent
    const float* down_mask[4]; const int* down_nbr[4];
    const float* maskR[5];     const int* nbrR[5];
    const float* onehot[4];    const int* parent[4];
    for (int l = 0; l < 4; ++l) down_mask[l] = f32p(idx++);
    for (int l = 0; l < 4; ++l) down_nbr[l]  = i32p(idx++);
    for (int l = 0; l < 5; ++l) maskR[l]     = f32p(idx++);
    int nbrBase = idx;
    for (int l = 0; l < 5; ++l) nbrR[l]      = i32p(idx++);
    for (int l = 0; l < 4; ++l) onehot[l]    = f32p(idx++);
    for (int l = 0; l < 4; ++l) parent[l]    = i32p(idx++);

    int NL[5];
    for (int l = 0; l < 5; ++l) NL[l] = in_sizes[nbrBase + l] / 27;

    // ---- workspace bump allocator (deterministic per call) ----
    size_t off = 0;
    auto alloc = [&](size_t bytes) -> void* {
        off = (off + 255) & ~(size_t)255;
        void* p = (char*)d_ws + off;
        off += bytes;
        return p;
    };

    // convert + transpose weight: W (Ktot x cout) -> Wt (cout x Ktot), bf16
    auto tobft = [&](const float* w, int Ktot, int cout) -> bf16* {
        size_t n = (size_t)Ktot * cout;
        bf16* p = (bf16*)alloc(n * sizeof(bf16));
        k_w_to_bf16_t<<<cdiv((int)n, 256), 256, 0, stream>>>(w, p, Ktot, cout);
        return p;
    };

    size_t maxNC = 0;
    for (int l = 0; l < 5; ++l) {
        size_t s = (size_t)NL[l] * (l < 4 ? 2 * NP[l] : NP[l]);
        if (s > maxNC) maxNC = s;
    }

    float* sums = (float*)alloc(512 * sizeof(float));
    float* sqs  = (float*)alloc(512 * sizeof(float));
    float* t1   = (float*)alloc(maxNC * sizeof(float));
    float* t2   = (float*)alloc(maxNC * sizeof(float));
    bf16*  bf1  = (bf16*)alloc(maxNC * sizeof(bf16));
    float* cat  = (float*)alloc(maxNC * sizeof(float));
    float* xb[5];
    for (int l = 0; l < 5; ++l)
        xb[l] = (float*)alloc((size_t)NL[l] * (l < 4 ? 2 * NP[l] : NP[l]) * sizeof(float));

    // ---- fused rulebooks (mask folded into index, -1 = inactive) ----
    int* nbrF[5]; int* downF[4];
    for (int l = 0; l < 5; ++l) {
        int n = 27 * NL[l];
        nbrF[l] = (int*)alloc((size_t)n * sizeof(int));
        k_fuse_rule<<<cdiv(n, 256), 256, 0, stream>>>(nbrR[l], maskR[l], nbrF[l], n);
    }
    for (int l = 0; l < 4; ++l) {
        int n = 8 * NL[l + 1];
        downF[l] = (int*)alloc((size_t)n * sizeof(int));
        k_fuse_rule<<<cdiv(n, 256), 256, 0, stream>>>(down_nbr[l], down_mask[l], downF[l], n);
    }

    // ---- launch helpers ----
    auto bnrelu = [&](const float* x, BN bn, int n, int c, bf16* o) {
        k_colsum<<<c, 256, 0, stream>>>(x, n, c, sums, sqs);
        k_bn_relu_bf16<<<cdiv(n * c, 256), 256, 0, stream>>>(x, sums, sqs, bn.g, bn.b, n, c, o);
    };
    auto gather = [&](const bf16* X, const bf16* Wt, const int* nbrm,
                      float* out, int Nout, int cin, int cout, int Koffs,
                      int ldc, int colOff, int accum) {
        int cin8 = cin / 8;
        unsigned mul16 = (65536u + (unsigned)cin8 - 1u) / (unsigned)cin8;
        dim3 g(cdiv(Nout, 64), cout / 16);
        k_gather_wmma<<<g, 128, 0, stream>>>(X, Wt, nbrm, out, Nout, cin, cin8,
                                             mul16, Koffs * cin, ldc, colOff, accum);
    };
    auto resblock = [&](float* x, const Block& bp, int N, int cin, int cout,
                        const int* nbrm) {
        bnrelu(x, bp.bn1, N, cin, bf1);
        gather(bf1, tobft(bp.conv1, 27 * cin, cout), nbrm,
               t1, N, cin, cout, 27, cout, 0, 0);
        bnrelu(t1, bp.bn2, N, cout, bf1);
        if (bp.nin) {
            gather(bf1, tobft(bp.conv2, 27 * cout, cout), nbrm,
                   t2, N, cout, cout, 27, cout, 0, 0);
            k_f32_to_bf16<<<cdiv(N * cin, 256), 256, 0, stream>>>(x, bf1, N * cin);
            gather(bf1, tobft(bp.nin, cin, cout), nullptr,
                   t2, N, cin, cout, 1, cout, 0, 1);
            k_copy_cols<<<cdiv(N * cout, 256), 256, 0, stream>>>(t2, x, N, cout, cout, cout, 0);
        } else {
            // identity shortcut: accumulate conv2 output directly into x
            gather(bf1, tobft(bp.conv2, 27 * cout, cout), nbrm,
                   x, N, cout, cout, 27, cout, 0, 1);
        }
    };

    // ---- input conv (cin=1, dedicated kernel): feats -> xb[0] (N0 x 16) ----
    k_in_conv<<<cdiv(NL[0] * 16, 256), 256, 0, stream>>>(feats, in_conv, nbrF[0],
                                                         xb[0], NL[0]);

    // ---- down path ----
    for (int l = 0; l < 4; ++l) {
        int N = NL[l], C = NP[l];
        resblock(xb[l], pre_b[l][0], N, C, C, nbrF[l]);
        resblock(xb[l], pre_b[l][1], N, C, C, nbrF[l]);
        bnrelu(xb[l], down_bn[l], N, C, bf1);
        gather(bf1, tobft(down_w[l], 8 * C, NP[l + 1]), downF[l],
               xb[l + 1], NL[l + 1], C, NP[l + 1], 8, NP[l + 1], 0, 0);
    }

    // ---- bottom ----
    resblock(xb[4], pre_b[4][0], NL[4], NP[4], NP[4], nbrF[4]);
    resblock(xb[4], pre_b[4][1], NL[4], NP[4], NP[4], nbrF[4]);

    // ---- up path ----
    for (int l = 3; l >= 0; --l) {
        int N = NL[l], C = NP[l], Nc = NL[l + 1], Cc = NP[l + 1];
        bnrelu(xb[l + 1], up_bn[l], Nc, Cc, bf1);
        k_copy_cols<<<cdiv(N * C, 256), 256, 0, stream>>>(xb[l], cat, N, C, C, 2 * C, 0);
        k_deconv<<<cdiv(N * C, 256), 256, 0, stream>>>(bf1, up_w[l], parent[l], onehot[l],
                                                       cat, N, Cc, C, 2 * C, C);
        resblock(cat, post_b[l][0], N, 2 * C, C, nbrF[l]);
        resblock(cat, post_b[l][1], N, C, C, nbrF[l]);
        k_copy_cols<<<cdiv(N * C, 256), 256, 0, stream>>>(cat, xb[l], N, C, C, C, 0);
    }

    // ---- head: bn_relu + linear 16 -> 5 ----
    bnrelu(xb[0], final_bn, NL[0], M16, bf1);
    k_linear<<<cdiv(NL[0] * 5, 256), 256, 0, stream>>>(bf1, lin_w, lin_b,
                                                       (float*)d_out, NL[0], M16, 5);
}
